// GCN_18433999635057
// MI455X (gfx1250) — compile-verified, matrix operand order checked by reading, not run on hardware
//
#include <hip/hip_runtime.h>
#include <hip/hip_bf16.h>

typedef __attribute__((ext_vector_type(8)))  __bf16 v8bf;
typedef __attribute__((ext_vector_type(16))) __bf16 v16bf;
typedef __attribute__((ext_vector_type(8)))  float  v8f;
typedef unsigned int u32x4 __attribute__((ext_vector_type(4)));
typedef unsigned int u32x8 __attribute__((ext_vector_type(8)));

#define N_NODES   100000
#define N_EDGES   1600000
#define D_HID     128
#define D_OUTF    40
#define NUM_CONVS 6
#define BN_EPS    1e-5f

__device__ __forceinline__ void atom_add_f32(float* p, float v) {
    __hip_atomic_fetch_add(p, v, __ATOMIC_RELAXED, __HIP_MEMORY_SCOPE_AGENT);
}

// ---------------- degree / norm ----------------
__global__ void __launch_bounds__(256) k_init_deg(float* deg, int n) {
    int i = blockIdx.x * 256 + threadIdx.x;
    if (i < n) deg[i] = 1.0f;   // self loop contributes 1
}

__global__ void __launch_bounds__(256) k_deg_edges(const int* __restrict__ ei, float* deg) {
    int e = blockIdx.x * 256 + threadIdx.x;
    if (e < N_EDGES) atom_add_f32(&deg[ei[N_EDGES + e]], 1.0f);
}

__global__ void __launch_bounds__(256) k_dinv(const float* __restrict__ deg, float* dinv, int n) {
    int i = blockIdx.x * 256 + threadIdx.x;
    if (i < n) dinv[i] = rsqrtf(deg[i]);
}

// ---------------- convert f32 -> bf16 ----------------
__global__ void __launch_bounds__(256) k_f32_to_bf16(const float* __restrict__ in,
                                                     __bf16* __restrict__ out, int n) {
    int i = blockIdx.x * 256 + threadIdx.x;
    if (i < n) out[i] = (__bf16)in[i];
}

// ---------------- weight packing into WMMA B-fragment layout ----------------
// Wp[((kt*Ntiles+nt)*32 + lane)*16 + e], lane 0-15 -> n=lane (K 0..7,16..23),
// lane 16-31 -> n=lane-16 (K 8..15,24..31); zero-pad out-of-range columns.
__global__ void __launch_bounds__(256) k_pack_w(const float* __restrict__ W,
                                                __bf16* __restrict__ Wp,
                                                int K, int ncols, int Ntiles, int total) {
    int idx = blockIdx.x * 256 + threadIdx.x;
    if (idx >= total) return;
    int e    = idx & 15;
    int lane = (idx >> 4) & 31;
    int tn   = idx >> 9;               // kt*Ntiles + nt
    int nt   = tn % Ntiles;
    int kt   = tn / Ntiles;
    int k = kt * 32 + ((e >> 3) << 4) + ((lane >> 4) << 3) + (e & 7);
    int n = nt * 16 + (lane & 15);
    float v = (k < K && n < ncols) ? W[k * ncols + n] : 0.0f;
    Wp[idx] = (__bf16)v;
}

// ---------------- WMMA GEMM: C[M x ncols] = A[M x 128](bf16) * Wp ----------------
// One wave per 16-row stripe, all NTILES output tiles per wave (A-fragment reuse).
// B is staged block-wide into LDS by the Tensor Data Mover (wave 0 issues, waits
// on TENSORcnt, then workgroup barrier).
template<int NTILES>
__global__ void __launch_bounds__(256) k_wmma_gemm(const __bf16* __restrict__ A,
                                                   const __bf16* __restrict__ Wp,
                                                   float* __restrict__ C,
                                                   int M, int ncols) {
    constexpr int K  = 128;
    constexpr int KT = K / 32;
    __shared__ __bf16 sW[KT * NTILES * 512];

    if ((threadIdx.x >> 5) == 0) {               // wave 0 only (wave-uniform branch)
        // ---- Tensor DMA descriptor (1-D tile of 8-byte elements) ----
        constexpr unsigned NB8 = (unsigned)(sizeof(sW) >> 3);   // # 8B elements
        const unsigned long long ga = (unsigned long long)(uintptr_t)Wp;
        const unsigned lds_off = (unsigned)(uintptr_t)(&sW[0]); // flat[31:0] == LDS offset
        u32x4 g0;
        g0[0] = 1u;                                             // count=1, user mode
        g0[1] = lds_off;                                        // lds_addr [63:32]
        g0[2] = (unsigned)ga;                                   // global_addr [95:64]
        g0[3] = ((unsigned)(ga >> 32) & 0x01FFFFFFu) | (2u << 30); // addr[56:32] | type=2
        u32x8 g1;
        g1[0] = 3u << 16;                                       // data_size=8B, mask=0
        g1[1] = (NB8 & 0xFFFFu) << 16;                          // tensor_dim0[15:0] @ bits63:48
        g1[2] = (NB8 >> 16) | (1u << 16);                       // tensor_dim0[31:16] | tensor_dim1=1
        g1[3] = (NB8 & 0xFFFFu) << 16;                          // tile_dim0 @ bits127:112
        g1[4] = 1u;                                             // tile_dim1=1, tile_dim2=0
        g1[5] = NB8;                                            // tensor_dim0_stride[31:0]
        g1[6] = 0u;
        g1[7] = 0u;
        asm volatile("tensor_load_to_lds %0, %1" :: "s"(g0), "s"(g1) : "memory");
        __builtin_amdgcn_s_wait_tensorcnt(0);
    }
    __syncthreads();

    const int lane = threadIdx.x & 31;
    const int wave = threadIdx.x >> 5;
    const int mt   = blockIdx.x * 8 + wave;
    if (mt * 16 >= M) return;                    // wave-uniform: EXEC all-ones below
    const int half = lane >> 4;
    const int lrow = lane & 15;
    const long arow = (long)(mt * 16 + lrow) * K;

    v8f acc[NTILES] = {};
#pragma unroll
    for (int kt = 0; kt < KT; ++kt) {
        union { v16bf v; v8bf h[2]; } a;
        const int kb = kt * 32 + half * 8;
        a.h[0] = *(const v8bf*)(A + arow + kb);
        a.h[1] = *(const v8bf*)(A + arow + kb + 16);
#pragma unroll
        for (int nt = 0; nt < NTILES; ++nt) {
            v16bf b = *(const v16bf*)(sW + ((size_t)(kt * NTILES + nt) * 32 + lane) * 16);
            acc[nt] = __builtin_amdgcn_wmma_f32_16x16x32_bf16(
                false, a.v, false, b, (short)0, acc[nt], false, false);
        }
    }

    const int rbase = mt * 16 + half * 8;
#pragma unroll
    for (int nt = 0; nt < NTILES; ++nt) {
        const int col = nt * 16 + lrow;
        if (col < ncols) {
            float* cp = C + (size_t)rbase * ncols + col;
#pragma unroll
            for (int v = 0; v < 8; ++v) cp[(size_t)v * ncols] = acc[nt][v];
        }
    }
}

// ---------------- aggregation: init with bias + self loop ----------------
__global__ void __launch_bounds__(256) k_agg_init(const float* __restrict__ hg,
                                                  const float* __restrict__ bias,
                                                  const float* __restrict__ dinv,
                                                  float* __restrict__ out, int D, int total) {
    int i = blockIdx.x * 256 + threadIdx.x;
    if (i >= total) return;
    int row = i / D, f = i - row * D;
    float di = dinv[row];
    out[i] = bias[f] + hg[i] * di * di;
}

// one wave per edge (D=128 -> 32 chunks of 4), coalesced float4 reads + 4 f32 atomics
__global__ void __launch_bounds__(256) k_agg_edges(const int* __restrict__ ei,
                                                   const float* __restrict__ hg,
                                                   const float* __restrict__ dinv,
                                                   float* __restrict__ out,
                                                   int D, int chunks, long total) {
    long tid = (long)blockIdx.x * 256 + threadIdx.x;
    if (tid >= total) return;
    int e = (int)(tid / chunks);
    int c = (int)(tid - (long)e * chunks);
    int s = ei[e];
    int d = ei[N_EDGES + e];
    float w = dinv[s] * dinv[d];
    const float4 v = *(const float4*)(hg + (size_t)s * D + c * 4);
    float* o = out + (size_t)d * D + c * 4;
    atom_add_f32(o + 0, v.x * w);
    atom_add_f32(o + 1, v.y * w);
    atom_add_f32(o + 2, v.z * w);
    atom_add_f32(o + 3, v.w * w);
}

// ---------------- batchnorm ----------------
__global__ void __launch_bounds__(256) k_zero_stats(float* stats) {
    stats[threadIdx.x] = 0.0f;
}

__global__ void __launch_bounds__(256) k_bn_stats(const float* __restrict__ h,
                                                  float* __restrict__ stats, int nrows) {
    __shared__ float s_sum[256], s_sq[256];
    int t = threadIdx.x;
    int f = t & 127;
    int half = t >> 7;
    float sum = 0.f, sq = 0.f;
    for (int r = blockIdx.x * 2 + half; r < nrows; r += gridDim.x * 2) {
        float v = h[(size_t)r * D_HID + f];
        sum += v; sq += v * v;
    }
    s_sum[t] = sum; s_sq[t] = sq;
    __syncthreads();
    if (t < 128) {
        atom_add_f32(&stats[f],       s_sum[t] + s_sum[t + 128]);
        atom_add_f32(&stats[128 + f], s_sq[t]  + s_sq[t + 128]);
    }
}

__global__ void __launch_bounds__(128) k_bn_finalize(const float* __restrict__ stats,
                                                     const float* __restrict__ gamma,
                                                     const float* __restrict__ beta,
                                                     float* __restrict__ sc, float inv_n) {
    int f = threadIdx.x;
    float mean = stats[f] * inv_n;
    float var  = stats[128 + f] * inv_n - mean * mean;
    float s    = gamma[f] * rsqrtf(var + BN_EPS);
    sc[f]       = s;
    sc[128 + f] = beta[f] - mean * s;
}

// BN apply + ReLU + f32->bf16 for next layer's GEMM input
__global__ void __launch_bounds__(256) k_bn_apply(const float* __restrict__ ha,
                                                  const float* __restrict__ sc,
                                                  __bf16* __restrict__ hb, int total) {
    int i = blockIdx.x * 256 + threadIdx.x;
    if (i >= total) return;
    int f = i & 127;
    float o = fmaxf(ha[i] * sc[f] + sc[128 + f], 0.0f);
    hb[i] = (__bf16)o;
}

// ---------------- host ----------------
extern "C" void kernel_launch(void* const* d_in, const int* in_sizes, int n_in,
                              void* d_out, int out_size, void* d_ws, size_t ws_size,
                              hipStream_t stream) {
    const float* x  = (const float*)d_in[0];
    const int*   ei = (const int*)d_in[1];           // edge_index [2, E]
    const float* Ws[NUM_CONVS];
    const float* bs[NUM_CONVS];
    const float* gammas[NUM_CONVS - 1];
    const float* betas[NUM_CONVS - 1];
    for (int i = 0; i < NUM_CONVS; ++i)     Ws[i]     = (const float*)d_in[2 + i];
    for (int i = 0; i < NUM_CONVS; ++i)     bs[i]     = (const float*)d_in[8 + i];
    for (int i = 0; i < NUM_CONVS - 1; ++i) gammas[i] = (const float*)d_in[14 + i];
    for (int i = 0; i < NUM_CONVS - 1; ++i) betas[i]  = (const float*)d_in[19 + i];

    char* p = (char*)d_ws;
    auto carve = [&](size_t bytes) {
        void* r = (void*)p;
        p += (bytes + 255) & ~(size_t)255;
        return r;
    };
    __bf16* hb   = (__bf16*)carve((size_t)N_NODES * D_HID * sizeof(__bf16));
    float*  hg   = (float*)carve((size_t)N_NODES * D_HID * sizeof(float));
    float*  ha   = (float*)carve((size_t)N_NODES * D_HID * sizeof(float));
    float*  deg  = (float*)carve((size_t)N_NODES * sizeof(float));
    float*  dinv = (float*)carve((size_t)N_NODES * sizeof(float));
    __bf16* Wp   = (__bf16*)carve((size_t)4 * 8 * 32 * 16 * sizeof(__bf16));
    float*  stats = (float*)carve(256 * sizeof(float));
    float*  bnp   = (float*)carve(256 * sizeof(float));
    (void)in_sizes; (void)n_in; (void)out_size; (void)ws_size;

    const int NB = (N_NODES + 255) / 256;
    k_init_deg<<<NB, 256, 0, stream>>>(deg, N_NODES);
    k_deg_edges<<<(N_EDGES + 255) / 256, 256, 0, stream>>>(ei, deg);
    k_dinv<<<NB, 256, 0, stream>>>(deg, dinv, N_NODES);

    k_f32_to_bf16<<<(N_NODES * D_HID + 255) / 256, 256, 0, stream>>>(x, hb, N_NODES * D_HID);

    const int Mtiles = N_NODES / 16;
    const int gemm_blocks = (Mtiles + 7) / 8;

    for (int l = 0; l < NUM_CONVS; ++l) {
        const int K      = D_HID;
        const int ncols  = (l == NUM_CONVS - 1) ? D_OUTF : D_HID;
        const int Ntiles = (ncols + 15) / 16;          // 8 or 3 (padded to 48)
        const int Kt     = K / 32;

        int pack_total = Kt * Ntiles * 512;
        k_pack_w<<<(pack_total + 255) / 256, 256, 0, stream>>>(Ws[l], Wp, K, ncols, Ntiles,
                                                               pack_total);

        if (l == NUM_CONVS - 1)
            k_wmma_gemm<3><<<gemm_blocks, 256, 0, stream>>>(hb, Wp, hg, N_NODES, ncols);
        else
            k_wmma_gemm<8><<<gemm_blocks, 256, 0, stream>>>(hb, Wp, hg, N_NODES, ncols);

        float* aggout = (l == NUM_CONVS - 1) ? (float*)d_out : ha;
        int tot = N_NODES * ncols;
        k_agg_init<<<(tot + 255) / 256, 256, 0, stream>>>(hg, bs[l], dinv, aggout, ncols, tot);

        int  chunks = ncols / 4;
        long etot   = (long)N_EDGES * chunks;
        k_agg_edges<<<(unsigned)((etot + 255) / 256), 256, 0, stream>>>(ei, hg, dinv, aggout,
                                                                        ncols, chunks, etot);

        if (l < NUM_CONVS - 1) {
            k_zero_stats<<<1, 256, 0, stream>>>(stats);
            k_bn_stats<<<1024, 256, 0, stream>>>(ha, stats, N_NODES);
            k_bn_finalize<<<1, 128, 0, stream>>>(stats, gammas[l], betas[l], bnp,
                                                 1.0f / (float)N_NODES);
            k_bn_apply<<<(N_NODES * D_HID + 255) / 256, 256, 0, stream>>>(ha, bnp, hb,
                                                                          N_NODES * D_HID);
        }
    }
}
